// Model_51556787421664
// MI455X (gfx1250) — matrix operand order repaired and checked by reading, not compiled, not run on hardware
//
#include <hip/hip_runtime.h>
#include <hip/hip_bf16.h>
#include <stdint.h>

typedef __bf16 bf16;
typedef __attribute__((ext_vector_type(16))) __bf16 v16bf;
typedef __attribute__((ext_vector_type(8)))  float  v8f;
typedef __attribute__((ext_vector_type(4)))  uint32_t u32x4;
typedef __attribute__((ext_vector_type(8)))  uint32_t u32x8;

#define N_PTS 4096
#define TD    8192   // T_IN * D_IN
#define F_DIM 2048   // target_length * output_dim

#define BM 128       // workgroup tile M
#define BN 128       // workgroup tile N
#define KC 64        // K chunk staged per LDS buffer

union Frag { v16bf v; uint4 q[2]; };

// ---------------------------------------------------------------------------
// Tensor Data Mover: one instruction DMAs a 2-D bf16 tile (tileD1 rows x
// tileD0 elements, row stride strideD0) from global memory into LDS.
// D# group0: count=1 | lds_addr | global_addr | type=2.
// D# group1: data_size=2B | tensor dims | tile dims | dim0 stride.
// 2-D form: VADDR2/VADDR3 omitted (NULL). Tracked by TENSORcnt.
// ---------------------------------------------------------------------------
__device__ __forceinline__ void tdm_load_2d(unsigned ldsAddr, const bf16* gbase,
                                            unsigned tensorD0, unsigned tensorD1,
                                            unsigned strideD0,
                                            unsigned tileD0, unsigned tileD1) {
    const uint64_t ga = (uint64_t)(uintptr_t)gbase;
    u32x4 g0;
    g0[0] = 1u;                                   // count=1, user descriptor
    g0[1] = ldsAddr;                              // LDS byte address
    g0[2] = (uint32_t)ga;                         // global_addr[31:0]
    g0[3] = (uint32_t)(ga >> 32) | (2u << 30);    // global_addr[56:32] | type=2
    u32x8 g1;
    g1[0] = 0x00010000u;                          // data_size = 2 bytes; no mask/pad
    g1[1] = (tensorD0 & 0xFFFFu) << 16;                          // tensor_dim0[15:0]
    g1[2] = (tensorD0 >> 16) | ((tensorD1 & 0xFFFFu) << 16);     // dim0 hi | dim1 lo
    g1[3] = (tensorD1 >> 16) | ((tileD0  & 0xFFFFu) << 16);      // dim1 hi | tile_dim0
    g1[4] = tileD1 & 0xFFFFu;                                    // tile_dim1 (tile_dim2=0)
    g1[5] = strideD0;                                            // tensor_dim0_stride lo
    g1[6] = 0u;                                                  // stride hi / dim1_stride
    g1[7] = 0u;
    asm volatile("tensor_load_to_lds %0, %1" :: "s"(g0), "s"(g1) : "memory");
}

// ---------------------------------------------------------------------------
// 128x128 output tile per 256-thread workgroup (8 waves, 4(M) x 2(N)).
// Wave w owns a 32x64 region = 2x4 subtiles of 16x16 (8 f32 accumulators).
// A[m, k] row-major (lda = K stride), Bt[n, k] row-major (ldb = K stride).
// K staged in chunks of 64 via TDM (each wave DMAs a 16-row slice of the A
// and B tiles), double-buffered LDS, TENSORcnt-gated.
// Per chunk per wave: 16 v_wmma : 24 ds_load_b128 : 2 tensor_load_to_lds.
// ---------------------------------------------------------------------------
template <int KDIM>
__device__ __forceinline__ void wg_gemm128(const bf16* __restrict__ A, int lda,
                                           const bf16* __restrict__ Bt, int ldb,
                                           int m0, int n0, v8f (&acc)[2][4]) {
    __shared__ bf16 lA[2][BM][KC];
    __shared__ bf16 lB[2][BN][KC];

    const int tid  = threadIdx.x;   // 256
    const int lane = tid & 31;
    const int wave = tid >> 5;

    // scalar wave id -> descriptor math stays in SALU
    const int wv = __builtin_amdgcn_readfirstlane(tid) >> 5;

    const unsigned ldsA_w = (unsigned)(uintptr_t)&lA[0][wv * 16][0];
    const unsigned ldsB_w = (unsigned)(uintptr_t)&lB[0][wv * 16][0];
    const bf16* gA_w = A  + (size_t)(m0 + wv * 16) * (unsigned)lda;
    const bf16* gB_w = Bt + (size_t)(n0 + wv * 16) * (unsigned)ldb;
    const unsigned bufA = (unsigned)(BM * KC * 2);   // bytes per buffer
    const unsigned bufB = (unsigned)(BN * KC * 2);

    // fragment addressing (contiguous k = 16*(lane>>4) + s, same for A and B)
    const int wm  = wave & 3;                  // M offset 32*wm
    const int wn  = wave >> 2;                 // N offset 64*wn
    const int rA0 = wm * 32 + (lane & 15);     // + 16*i for subtile row
    const int kAe = (lane >> 4) * 16;          // A: contiguous run {kAe..+15}
    const int rB0 = wn * 64 + (lane & 15);     // + 16*j for subtile col
    const int kBe = (lane >> 4) * 16;          // B: contiguous run {kBe..+15}

#pragma unroll
    for (int i = 0; i < 2; ++i)
#pragma unroll
        for (int j = 0; j < 4; ++j)
            acc[i][j] = (v8f){0,0,0,0,0,0,0,0};

    auto issue = [&](int buf, int kc) {
        tdm_load_2d(ldsA_w + (unsigned)buf * bufA, gA_w + kc,
                    (unsigned)lda, 1u << 20, (unsigned)lda, KC, 16);
        tdm_load_2d(ldsB_w + (unsigned)buf * bufB, gB_w + kc,
                    (unsigned)ldb, 1u << 20, (unsigned)ldb, KC, 16);
    };

    issue(0, 0);   // prologue: every wave stages its slice of chunk 0

    int buf = 0;
#pragma unroll 2
    for (int kc = 0; kc < KDIM; kc += KC) {
        const bool hasNext = (kc + KC) < KDIM;
        if (hasNext) {
            issue(buf ^ 1, kc + KC);
            __builtin_amdgcn_s_wait_tensorcnt(2);   // this wave's current slice landed
        } else {
            __builtin_amdgcn_s_wait_tensorcnt(0);
        }
        __syncthreads();   // publish all waves' slices

#pragma unroll
        for (int kk = 0; kk < KC; kk += 32) {
            Frag fa[2], fb[4];
#pragma unroll
            for (int i = 0; i < 2; ++i) {
                fa[i].q[0] = *(const uint4*)&lA[buf][rA0 + 16 * i][kk + kAe];
                fa[i].q[1] = *(const uint4*)&lA[buf][rA0 + 16 * i][kk + kAe + 8];
            }
#pragma unroll
            for (int j = 0; j < 4; ++j) {
                fb[j].q[0] = *(const uint4*)&lB[buf][rB0 + 16 * j][kk + kBe];
                fb[j].q[1] = *(const uint4*)&lB[buf][rB0 + 16 * j][kk + kBe + 8];
            }
#pragma unroll
            for (int i = 0; i < 2; ++i)
#pragma unroll
                for (int j = 0; j < 4; ++j)
                    acc[i][j] = __builtin_amdgcn_wmma_f32_16x16x32_bf16(
                        false, fa[i].v, false, fb[j].v, (short)0, acc[i][j], false, false);
        }
        __syncthreads();   // compute done before next chunk overwrites this buffer
        buf ^= 1;
    }
}

// ---------------------------------------------------------------------------
// Stage 1a: f32 -> bf16 elementwise (xs)
// ---------------------------------------------------------------------------
__global__ void __launch_bounds__(256) k_cvt(const float* __restrict__ in,
                                             bf16* __restrict__ out, int n4) {
    int i = blockIdx.x * 256 + threadIdx.x;
    if (i < n4) {
        float4 f = ((const float4*)in)[i];
        union { bf16 h[4]; unsigned long long u; } o;
        o.h[0] = (bf16)f.x; o.h[1] = (bf16)f.y; o.h[2] = (bf16)f.z; o.h[3] = (bf16)f.w;
        ((unsigned long long*)out)[i] = o.u;
    }
}

// ---------------------------------------------------------------------------
// Stage 1b: Wt[n][k] = (bf16) W[k][n]   (32x32 LDS-tiled transpose)
// ---------------------------------------------------------------------------
__global__ void __launch_bounds__(256) k_wt(const float* __restrict__ W,
                                            bf16* __restrict__ Wt) {
    __shared__ float tile[32][33];
    const int nblk = blockIdx.x * 32;   // over F_DIM
    const int kblk = blockIdx.y * 32;   // over TD
    const int tx = threadIdx.x, ty = threadIdx.y;  // (32, 8)
#pragma unroll
    for (int s = 0; s < 4; ++s)
        tile[ty + 8 * s][tx] = W[(size_t)(kblk + ty + 8 * s) * F_DIM + (nblk + tx)];
    __syncthreads();
#pragma unroll
    for (int s = 0; s < 4; ++s)
        Wt[(size_t)(nblk + ty + 8 * s) * TD + (kblk + tx)] = (bf16)tile[tx][ty + 8 * s];
}

// ---------------------------------------------------------------------------
// Stage 2: X = xs_b @ Wt^T  (i.e. xs @ W), store bf16
// ---------------------------------------------------------------------------
__global__ void __launch_bounds__(256) k_gemm_x(const bf16* __restrict__ A,
                                                const bf16* __restrict__ Bt,
                                                bf16* __restrict__ X) {
    const int m0 = blockIdx.x * BM;
    const int n0 = blockIdx.y * BN;
    v8f acc[2][4];
    wg_gemm128<TD>(A, TD, Bt, TD, m0, n0, acc);

    const int lane = threadIdx.x & 31;
    const int wave = threadIdx.x >> 5;
    const int wm = wave & 3, wn = wave >> 2;
    const int mB = m0 + 32 * wm + ((lane >> 4) << 3);  // C layout: lanes 16-31 hold M+8
    const int nB = n0 + 64 * wn + (lane & 15);
#pragma unroll
    for (int i = 0; i < 2; ++i)
#pragma unroll
        for (int j = 0; j < 4; ++j)
#pragma unroll
            for (int r = 0; r < 8; ++r)
                X[(size_t)(mB + 16 * i + r) * F_DIM + nB + 16 * j] = (bf16)acc[i][j][r];
}

// ---------------------------------------------------------------------------
// Stage 3: sq[i] = || X[i,:] ||^2
// ---------------------------------------------------------------------------
__global__ void __launch_bounds__(256) k_sq(const bf16* __restrict__ X,
                                            float* __restrict__ sq) {
    const int row = blockIdx.x;
    float s = 0.f;
    for (int c = threadIdx.x; c < F_DIM; c += 256) {
        float x = (float)X[(size_t)row * F_DIM + c];
        s += x * x;
    }
    for (int o = 16; o; o >>= 1) s += __shfl_xor(s, o, 32);
    __shared__ float ps[8];
    if ((threadIdx.x & 31) == 0) ps[threadIdx.x >> 5] = s;
    __syncthreads();
    if (threadIdx.x == 0) {
        float t = 0.f;
        for (int w = 0; w < 8; ++w) t += ps[w];
        sq[row] = t;
    }
}

__global__ void k_init(float* accum) {
    if (threadIdx.x == 0 && blockIdx.x == 0) *accum = 0.f;
}

// ---------------------------------------------------------------------------
// Stage 4: gram tiles (upper-triangular 128x128 block pairs only), fused
// d2 -> exp -> y_i y_j a_i a_j weighting -> scalar reduction.
// ---------------------------------------------------------------------------
__global__ void __launch_bounds__(256) k_gram(const bf16* __restrict__ X,
                                              const float* __restrict__ sq,
                                              const int* __restrict__ ys,
                                              const float* __restrict__ alphas,
                                              float* __restrict__ accum) {
    const int NB = N_PTS / BM;   // 32
    int t = blockIdx.x, bi = 0;
    while (t >= NB - bi) { t -= NB - bi; ++bi; }
    const int bj = bi + t;
    const int i0 = bi * BM, j0 = bj * BN;

    v8f acc[2][4];
    wg_gemm128<F_DIM>(X, F_DIM, X, F_DIM, i0, j0, acc);

    const int lane = threadIdx.x & 31;
    const int wave = threadIdx.x >> 5;
    const int wm = wave & 3, wn = wave >> 2;
    const int iBase = i0 + 32 * wm + ((lane >> 4) << 3);
    const int jBase = j0 + 64 * wn + (lane & 15);
    const float gamma = 1.0f / (float)F_DIM;

    float wi[2][8], si[2][8];
#pragma unroll
    for (int i = 0; i < 2; ++i)
#pragma unroll
        for (int r = 0; r < 8; ++r) {
            int idx = iBase + 16 * i + r;
            wi[i][r] = (float)(2 * ys[idx] - 1) * alphas[idx];
            si[i][r] = sq[idx];
        }
    float wj[4], sj[4];
#pragma unroll
    for (int j = 0; j < 4; ++j) {
        int jdx = jBase + 16 * j;
        wj[j] = (float)(2 * ys[jdx] - 1) * alphas[jdx];
        sj[j] = sq[jdx];
    }

    float lsum = 0.f;
#pragma unroll
    for (int i = 0; i < 2; ++i)
#pragma unroll
        for (int j = 0; j < 4; ++j)
#pragma unroll
            for (int r = 0; r < 8; ++r) {
                float d2 = fmaxf(si[i][r] + sj[j] - 2.0f * acc[i][j][r], 0.f);
                lsum += wi[i][r] * wj[j] * __expf(-gamma * d2);
            }
    for (int o = 16; o; o >>= 1) lsum += __shfl_xor(lsum, o, 32);
    if (lane == 0) {
        const float f = (bi == bj) ? 1.0f : 2.0f;   // symmetry: off-diag blocks x2
        atomicAdd(accum, f * lsum);
    }
}

// ---------------------------------------------------------------------------
// Stage 5: out = 0.5 * acc - sum(alphas)
// ---------------------------------------------------------------------------
__global__ void __launch_bounds__(256) k_final(const float* __restrict__ accum,
                                               const float* __restrict__ alphas,
                                               float* __restrict__ out) {
    float s = 0.f;
    for (int i = threadIdx.x; i < N_PTS; i += 256) s += alphas[i];
    for (int o = 16; o; o >>= 1) s += __shfl_xor(s, o, 32);
    __shared__ float ps[8];
    if ((threadIdx.x & 31) == 0) ps[threadIdx.x >> 5] = s;
    __syncthreads();
    if (threadIdx.x == 0) {
        float tot = 0.f;
        for (int w = 0; w < 8; ++w) tot += ps[w];
        out[0] = 0.5f * (*accum) - tot;
    }
}

extern "C" void kernel_launch(void* const* d_in, const int* in_sizes, int n_in,
                              void* d_out, int out_size, void* d_ws, size_t ws_size,
                              hipStream_t stream) {
    (void)in_sizes; (void)n_in; (void)out_size; (void)ws_size;
    const float* xs = (const float*)d_in[0];
    const float* W  = (const float*)d_in[1];
    const int*   ys = (const int*)d_in[2];
    const float* al = (const float*)d_in[3];
    float* out = (float*)d_out;

    char* ws = (char*)d_ws;
    bf16*  xs_b = (bf16*) (ws);                                   // 64 MB
    bf16*  Wt_b = (bf16*) (ws + ((size_t)64 << 20));              // 32 MB
    bf16*  X_b  = (bf16*) (ws + ((size_t)96 << 20));              // 16 MB
    float* sq   = (float*)(ws + ((size_t)112 << 20));             // 16 KB
    float* acc  = (float*)(ws + ((size_t)112 << 20) + (1 << 16)); // 4 B

    // 1) precision conversion + W transpose (one-time, bandwidth-trivial)
    k_cvt<<<(N_PTS * TD / 4) / 256, 256, 0, stream>>>(xs, xs_b, N_PTS * TD / 4);
    k_wt<<<dim3(F_DIM / 32, TD / 32), dim3(32, 8), 0, stream>>>(W, Wt_b);

    // 2) encoder GEMM: X = xs @ W   (137 GFLOP, bf16 WMMA, TDM-staged)
    k_gemm_x<<<dim3(N_PTS / BM, F_DIM / BN), 256, 0, stream>>>(xs_b, Wt_b, X_b);

    // 3) row norms
    k_sq<<<N_PTS, 256, 0, stream>>>(X_b, sq);

    // 4) fused symmetric gram + RBF + quadratic form reduction
    k_init<<<1, 64, 0, stream>>>(acc);
    const int NB = N_PTS / BM;   // 32
    k_gram<<<NB * (NB + 1) / 2, 256, 0, stream>>>(X_b, sq, ys, al, acc);

    // 5) finalize scalar
    k_final<<<1, 256, 0, stream>>>(acc, al, out);
}
